// Convolucion_Transformable_1443109012288
// MI455X (gfx1250) — matrix operand order repaired and checked by reading, not compiled, not run on hardware
//
#include <hip/hip_runtime.h>
#include <hip/hip_bf16.h>

// ---- problem constants (from reference setup_inputs) ----
#define B_    8
#define CIN   384
#define COUT  384
#define NLEN  4096
#define MU    3
#define KDIM  (CIN*MU)     // 1152
#define NPOS  (B_*NLEN)    // 32768

typedef __attribute__((ext_vector_type(16))) __bf16 v16bf;
typedef __attribute__((ext_vector_type(8)))  float  v8f;

__device__ __forceinline__ unsigned short f2bf(float f) {
    unsigned u = __float_as_uint(f);
    u += 0x7FFFu + ((u >> 16) & 1u);   // round-to-nearest-even
    return (unsigned short)(u >> 16);
}

// ---------------- kernel 1: W fp32 -> bf16 ----------------
__global__ void k_convert_w(const float* __restrict__ w,
                            unsigned short* __restrict__ wb, int n) {
    int i = blockIdx.x * blockDim.x + threadIdx.x;
    if (i < n) wb[i] = f2bf(w[i]);
}

// ---------------- kernel 2: offsets / modulation ----------------
__global__ __launch_bounds__(256) void k_coef(
    const float* __restrict__ x,
    const float* __restrict__ wdw, const float* __restrict__ wm,
    int2* __restrict__ cIdx, float2* __restrict__ cW)
{
    __shared__ float s_wd[MU * CIN * MU];
    __shared__ float s_wm[MU * CIN * MU];
    for (int i = threadIdx.x; i < MU * CIN * MU; i += 256) {
        s_wd[i] = wdw[i];
        s_wm[i] = wm[i];
    }
    __syncthreads();

    int pos = blockIdx.x * 256 + threadIdx.x;      // block never straddles b
    int b = pos / NLEN, n = pos % NLEN;
    const float* xb = x + (size_t)b * CIN * NLEN;

    float dwd[3] = {0.f, 0.f, 0.f};
    float md[3]  = {0.f, 0.f, 0.f};
    for (int c = 0; c < CIN; ++c) {
        const float* xc = xb + (size_t)c * NLEN + n;
        float xm1 = (n > 0)        ? xc[-1] : 0.f;   // zero pad
        float x0  = xc[0];
        float xp1 = (n < NLEN - 1) ? xc[1]  : 0.f;
#pragma unroll
        for (int j = 0; j < 3; ++j) {
            const float* wd = &s_wd[(j * CIN + c) * MU];
            const float* wq = &s_wm[(j * CIN + c) * MU];
            dwd[j] += xm1 * wd[0] + x0 * wd[1] + xp1 * wd[2];
            md[j]  += xm1 * wq[0] + x0 * wq[1] + xp1 * wq[2];
        }
    }
#pragma unroll
    for (int m = 0; m < 3; ++m) {
        float d  = dwd[m];
        int   ip = (int)d;                 // trunc toward zero == astype(int32)
        float fr = d - (float)ip;
        int g0 = n + (m - 1) + ip;
        int g1 = g0 + (fr >= 0.f ? 1 : -1);
        float c1 = (g0 >= 0 && g0 < NLEN) ? 1.f : 0.f;
        float c2 = (g1 >= 0 && g1 < NLEN) ? 1.f : 0.f;
        int g0c = min(max(g0, 0), NLEN - 1);
        int g1c = min(max(g1, 0), NLEN - 1);
        float af = fabsf(fr);
        float sg = 1.f / (1.f + __expf(-md[m]));
        cIdx[pos * 3 + m] = make_int2(g0c, g1c);
        cW[pos * 3 + m]   = make_float2((1.f - af) * c1 * sg, af * c2 * sg);
    }
}

// ---------------- kernel 3: WMMA GEMM ----------------
// Block = 256 threads = 8 waves, 64-position tile (4 B-tiles) per block.
// Each wave owns 3 o-tiles (wave, wave+8, wave+16) x 4 pos-tiles
// -> 12 v_wmma_f32_16x16x32_bf16 per K-step; A frags reused x4, B frags x3.
// v-tile double-buffered in LDS: one __syncthreads per K-step.
__global__ __launch_bounds__(256) void k_gemm(
    const float* __restrict__ x,
    const unsigned short* __restrict__ wb,   // bf16 W [COUT][KDIM]
    const float* __restrict__ bias,          // [COUT]
    const int2* __restrict__ cIdx, const float2* __restrict__ cW,
    float* __restrict__ y)
{
    __shared__ unsigned int s_v32[2][64 * 16]; // v-tile: [pos][k/2] dwords, x2 buffers
    __shared__ int2   s_idx[64 * 3];
    __shared__ float2 s_wt[64 * 3];

    const int tid  = threadIdx.x;
    const int wave = tid >> 5;
    const int lane = tid & 31;
    const int posbase = blockIdx.x * 64;       // 4096 % 64 == 0: same b
    const int b     = posbase / NLEN;
    const int nbase = posbase % NLEN;
    const float* xb = x + (size_t)b * CIN * NLEN;

    if (tid < 192) {
        s_idx[tid] = cIdx[posbase * 3 + tid];
        s_wt[tid]  = cW[posbase * 3 + tid];
    }
    __syncthreads();

    v8f acc[3][4];
#pragma unroll
    for (int o = 0; o < 3; ++o)
#pragma unroll
        for (int t = 0; t < 4; ++t)
            acc[o][t] = (v8f){};

    const int row = lane & 15;
    const int hi  = (lane >> 4) & 1;

    // v-tile fill ownership: thread -> position vp = tid/4, k-span = (tid%4)*8
    const int vp  = tid >> 2;
    const int vk0 = (tid & 3) << 3;

    union U { v16bf v; unsigned u[8]; };

    int buf = 0;
    for (int kk = 0; kk < KDIM; kk += 32) {
        // ---- build bf16 v tile (gather + lerp) ----
#pragma unroll
        for (int q = 0; q < 4; ++q) {
            unsigned out = 0;
#pragma unroll
            for (int h = 0; h < 2; ++h) {
                int k = kk + vk0 + 2 * q + h;
                int c = k / 3, m = k - 3 * c;
                int2   gi = s_idx[vp * 3 + m];
                float2 ww = s_wt[vp * 3 + m];
                const float* xc = xb + (size_t)c * NLEN;
                float v = ww.x * xc[gi.x] + ww.y * xc[gi.y];
                out |= ((unsigned)f2bf(v)) << (16 * h);
            }
            s_v32[buf][vp * 16 + (vk0 >> 1) + q] = out;
        }
        __syncthreads();

        // ---- 4 B fragments (32x16 each) from LDS ----
        U bfrag[4];
#pragma unroll
        for (int t = 0; t < 4; ++t)
#pragma unroll
            for (int p = 0; p < 8; ++p)
                bfrag[t].u[p] = s_v32[buf][(t * 16 + row) * 16 + (hi ? 8 : 0) + p];

        // ---- 3 A fragments (16x32 each) from bf16 W in L2 ----
        U afrag[3];
#pragma unroll
        for (int o = 0; o < 3; ++o) {
            int orow = (wave + 8 * o) * 16 + row;
#pragma unroll
            for (int p = 0; p < 8; ++p) {
                int kl = kk + (hi ? 16 : 0) + (p << 1);
                afrag[o].u[p] = *(const unsigned*)&wb[(size_t)orow * KDIM + kl];
            }
        }

        // ---- 12 WMMAs: full A/B fragment reuse ----
#pragma unroll
        for (int o = 0; o < 3; ++o)
#pragma unroll
            for (int t = 0; t < 4; ++t)
                acc[o][t] = __builtin_amdgcn_wmma_f32_16x16x32_bf16(
                    false, afrag[o].v, false, bfrag[t].v,
                    (short)0, acc[o][t], false, false);

        buf ^= 1;   // double buffer: next fill targets the other half
    }

    // ---- epilogue: bias + store (C layout: M = r + 8*hi, N = lane%16) ----
#pragma unroll
    for (int o = 0; o < 3; ++o) {
#pragma unroll
        for (int r = 0; r < 8; ++r) {
            int orow = (wave + 8 * o) * 16 + r + (hi ? 8 : 0);
            float bv = bias[orow];
            size_t base = ((size_t)b * COUT + orow) * NLEN;
#pragma unroll
            for (int t = 0; t < 4; ++t) {
                int ncol = nbase + t * 16 + row;
                y[base + ncol] = acc[o][t][r] + bv;
            }
        }
    }
}

extern "C" void kernel_launch(void* const* d_in, const int* in_sizes, int n_in,
                              void* d_out, int out_size, void* d_ws, size_t ws_size,
                              hipStream_t stream) {
    const float* x    = (const float*)d_in[0];   // (8,384,4096)
    const float* w    = (const float*)d_in[1];   // (384,384,3)
    const float* bias = (const float*)d_in[2];   // (1,384,1)
    const float* wdw  = (const float*)d_in[3];   // (3,384,3)
    const float* wm   = (const float*)d_in[4];   // (3,384,3)
    float* y = (float*)d_out;                    // (8,384,4096)

    unsigned char* ws = (unsigned char*)d_ws;
    unsigned short* wb = (unsigned short*)ws;                      // 884736 B
    int2*   cIdx = (int2*)(ws + 884736);                           // 786432 B
    float2* cW   = (float2*)(ws + 884736 + 786432);                // 786432 B

    k_convert_w<<<(COUT * KDIM + 255) / 256, 256, 0, stream>>>(w, wb, COUT * KDIM);
    k_coef<<<NPOS / 256, 256, 0, stream>>>(x, wdw, wm, cIdx, cW);
    k_gemm<<<NPOS / 64, 256, 0, stream>>>(x, wb, bias, cIdx, cW, y);
}
